// MultiHeadAttention_47674136986087
// MI455X (gfx1250) — compile-verified, matrix operand order checked by reading, not run on hardware
//
#include <hip/hip_runtime.h>
#include <stdint.h>

#define EMBED 1024
#define HEADS 16
#define HDIM  64
#define BATCH 4
#define SEQ   2048
#define ROWS  (BATCH * SEQ)     // 8192
#define QKV_N (3 * EMBED)       // 3072

typedef __bf16 bf16_t;
typedef bf16_t v16bf __attribute__((ext_vector_type(16)));
typedef float  v8f   __attribute__((ext_vector_type(8)));
typedef int    v4i   __attribute__((ext_vector_type(4)));

union FragA { v16bf v; unsigned int u[8]; };
union FragC { v8f  v; float f[8]; };

__device__ __forceinline__ unsigned short f2bf(float x) {
    union { float f; unsigned int u; } c; c.f = x;
    unsigned int r = (c.u + 0x7fffu + ((c.u >> 16) & 1u)) >> 16;  // RNE
    return (unsigned short)r;
}

// A-matrix 16x32 bf16 VGPR layout: dword j of lane (g = lane>>4) holds
// K = (j>=4 ? 16 : 0) + 8*g + 2*(j&3) .. +1  (ISA 7.12.2). B assumed symmetric.
__device__ __forceinline__ int kpair_of(int j, int g) {
    return ((j >> 2) << 4) | (g << 3) | ((j & 3) << 1);
}

// ---- async global->LDS copy (CDNA5 GLOBAL_LOAD_ASYNC_TO_LDS_B128, ASYNCcnt) ----
#if __has_builtin(__builtin_amdgcn_global_load_async_to_lds_b128)
#define HAVE_ASYNC_LDS 1
#else
#define HAVE_ASYNC_LDS 0
#endif

__device__ __forceinline__ void async_copy16(const void* gsrc, void* lds) {
#if HAVE_ASYNC_LDS
    __builtin_amdgcn_global_load_async_to_lds_b128(
        (__attribute__((address_space(1))) v4i*)gsrc,
        (__attribute__((address_space(3))) v4i*)lds, 0, 0);
#else
    *(uint4*)lds = *(const uint4*)gsrc;
#endif
}

__device__ __forceinline__ void wait_async0() {
#if HAVE_ASYNC_LDS
#if __has_builtin(__builtin_amdgcn_s_wait_asynccnt)
    __builtin_amdgcn_s_wait_asynccnt(0);
#else
    asm volatile("s_wait_asynccnt 0" ::: "memory");
#endif
#endif
}

// ---------------------------------------------------------------- fp32 -> bf16
__global__ void cvt_f32_bf16(const float* __restrict__ src,
                             unsigned short* __restrict__ dst, int n) {
    int i = blockIdx.x * blockDim.x + threadIdx.x;
    if (i < n) dst[i] = f2bf(src[i]);
}

// -------------------------------------------------- tiled bf16 GEMM + bias
// C[M,N] = A[M,K] @ B[K,N] + bias[N]; 128x128 tile, BK=32, 8 waves,
// each wave computes 32x64 (2x4 tiles of 16x16) via v_wmma_f32_16x16x32_bf16.
template <bool STORE_BF16>
__global__ __launch_bounds__(256)
void gemm_bf16_wmma(const unsigned short* __restrict__ A,
                    const unsigned short* __restrict__ Bm,
                    const float* __restrict__ bias,
                    void* __restrict__ Cout,
                    int M, int N, int K) {
    __shared__ unsigned short As[128][32];    // [m][k] row-major
    __shared__ unsigned short Bst[128][32];   // B transposed: [n][k]

    const int tid  = threadIdx.x;
    const int wid  = tid >> 5, lane = tid & 31;
    const int g    = lane >> 4, ln = lane & 15;
    const int wm   = wid >> 1,  wn = wid & 1;
    const int m0   = blockIdx.y * 128, n0 = blockIdx.x * 128;

    FragC acc[2][4];
#pragma unroll
    for (int tm = 0; tm < 2; ++tm)
#pragma unroll
        for (int tn = 0; tn < 4; ++tn)
#pragma unroll
            for (int r = 0; r < 8; ++r) acc[tm][tn].f[r] = 0.f;

    for (int kb = 0; kb < K; kb += 32) {
        __syncthreads();
        // A tile: straight row copy -> async global->LDS (2 x b128 per thread)
#pragma unroll
        for (int i = 0; i < 2; ++i) {
            int q = tid + i * 256;              // 0..511
            int row = q >> 2, seg = (q & 3) << 3;   // seg in elements (8 = 16B)
            async_copy16(A + (size_t)(m0 + row) * K + kb + seg, &As[row][seg]);
        }
        // B tile: transpose into LDS (needs VALU path)
#pragma unroll
        for (int i = 0; i < 8; ++i) {
            int q = tid + i * 256;
            int kk = q >> 6, n2 = (q & 63) << 1;
            unsigned int u = *(const unsigned int*)(Bm + (size_t)(kb + kk) * N + n0 + n2);
            Bst[n2][kk]     = (unsigned short)(u & 0xffffu);
            Bst[n2 + 1][kk] = (unsigned short)(u >> 16);
        }
        wait_async0();
        __syncthreads();

        FragA a[2], b[4];
#pragma unroll
        for (int tm = 0; tm < 2; ++tm)
#pragma unroll
            for (int j = 0; j < 8; ++j)
                a[tm].u[j] = *(const unsigned int*)&As[wm * 32 + tm * 16 + ln][kpair_of(j, g)];
#pragma unroll
        for (int tn = 0; tn < 4; ++tn)
#pragma unroll
            for (int j = 0; j < 8; ++j)
                b[tn].u[j] = *(const unsigned int*)&Bst[wn * 64 + tn * 16 + ln][kpair_of(j, g)];
#pragma unroll
        for (int tm = 0; tm < 2; ++tm)
#pragma unroll
            for (int tn = 0; tn < 4; ++tn)
                acc[tm][tn].v = __builtin_amdgcn_wmma_f32_16x16x32_bf16(
                    false, a[tm].v, false, b[tn].v, (short)0, acc[tm][tn].v, false, false);
    }

#pragma unroll
    for (int tm = 0; tm < 2; ++tm)
#pragma unroll
        for (int tn = 0; tn < 4; ++tn)
#pragma unroll
            for (int r = 0; r < 8; ++r) {
                int row = m0 + wm * 32 + tm * 16 + r + 8 * g;
                int col = n0 + wn * 64 + tn * 16 + ln;
                float v = acc[tm][tn].f[r] + bias[col];
                if (STORE_BF16)
                    ((unsigned short*)Cout)[(size_t)row * N + col] = f2bf(v);
                else
                    ((float*)Cout)[(size_t)row * N + col] = v;
            }
}

// ------------------------------------------------ flash-attention (streaming)
// grid = B*H*(SEQ/64), 128 threads (4 waves). Wave = 16 query rows.
__global__ __launch_bounds__(128)
void attention_wmma(const unsigned short* __restrict__ qkvb,   // [ROWS, 3E] bf16
                    const int* __restrict__ mask,              // [B*SEQ]
                    unsigned short* __restrict__ ctxb) {       // [ROWS, E] bf16
    __shared__ unsigned short Kt[64][64];      // [key][d]
    __shared__ unsigned short Vt[64][64];      // transposed: [d][key]
    __shared__ unsigned short Ps[4][16][64];   // per-wave P staging [qrow][key]

    const int tid = threadIdx.x;
    const int wid = tid >> 5, lane = tid & 31;
    const int g   = lane >> 4, ln = lane & 15;

    const int nqb = SEQ / 64;                       // 32
    const int qb  = blockIdx.x % nqb;
    const int h   = (blockIdx.x / nqb) % HEADS;
    const int b   = blockIdx.x / (nqb * HEADS);

    const int    q0   = qb * 64 + wid * 16;
    const size_t rowB = (size_t)b * SEQ;
    const float  scale = 0.125f;                    // 1/sqrt(64)

    // Q fragments (A-matrix 16x32, two for d=64) loaded once from global
    FragA aq[2];
#pragma unroll
    for (int f = 0; f < 2; ++f)
#pragma unroll
        for (int j = 0; j < 8; ++j) {
            int col = h * HDIM + f * 32 + kpair_of(j, g);
            aq[f].u[j] = *(const unsigned int*)(qkvb + (rowB + q0 + ln) * (size_t)QKV_N + col);
        }

    float m[8], l[8];
    FragC o[4];
#pragma unroll
    for (int r = 0; r < 8; ++r) { m[r] = -3.0e38f; l[r] = 0.f; }
#pragma unroll
    for (int dt = 0; dt < 4; ++dt)
#pragma unroll
        for (int r = 0; r < 8; ++r) o[dt].f[r] = 0.f;

    for (int kb = 0; kb < SEQ; kb += 64) {
        __syncthreads();
        // K tile: straight row copy -> async global->LDS (4 x b128 per thread)
#pragma unroll
        for (int i = 0; i < 4; ++i) {
            int q = tid + i * 128;                 // 0..511
            int key = q >> 3, seg = (q & 7) << 3;  // seg in elements (8 = 16B)
            async_copy16(qkvb + (rowB + kb + key) * (size_t)QKV_N + EMBED + h * HDIM + seg,
                         &Kt[key][seg]);
        }
        // V tile: transposed into LDS (VALU path)
#pragma unroll
        for (int i = 0; i < 16; ++i) {
            int q = tid + i * 128;
            int key = q >> 5, dp = (q & 31) << 1;
            unsigned int uv = *(const unsigned int*)(
                qkvb + (rowB + kb + key) * (size_t)QKV_N + 2 * EMBED + h * HDIM + dp);
            Vt[dp][key]     = (unsigned short)(uv & 0xffffu);
            Vt[dp + 1][key] = (unsigned short)(uv >> 16);
        }
        wait_async0();
        __syncthreads();

        // S = scale * Q K^T   (16 x 64 per wave), + mask
        FragC s[4];
#pragma unroll
        for (int n = 0; n < 4; ++n) {
#pragma unroll
            for (int r = 0; r < 8; ++r) s[n].f[r] = 0.f;
#pragma unroll
            for (int ks = 0; ks < 2; ++ks) {
                FragA bk;
#pragma unroll
                for (int j = 0; j < 8; ++j)
                    bk.u[j] = *(const unsigned int*)&Kt[n * 16 + ln][ks * 32 + kpair_of(j, g)];
                s[n].v = __builtin_amdgcn_wmma_f32_16x16x32_bf16(
                    false, aq[ks].v, false, bk.v, (short)0, s[n].v, false, false);
            }
            int mv = mask[b * SEQ + kb + n * 16 + ln];   // lane's column is fixed
#pragma unroll
            for (int r = 0; r < 8; ++r)
                s[n].f[r] = (mv == 0) ? -3.0e38f : s[n].f[r] * scale;
        }

        // online softmax: row max / sum live in the C-layout (r, half-g) striping
        float rm[8], rs[8];
#pragma unroll
        for (int r = 0; r < 8; ++r)
            rm[r] = fmaxf(fmaxf(s[0].f[r], s[1].f[r]), fmaxf(s[2].f[r], s[3].f[r]));
#pragma unroll
        for (int off = 8; off >= 1; off >>= 1)
#pragma unroll
            for (int r = 0; r < 8; ++r)
                rm[r] = fmaxf(rm[r], __shfl_xor(rm[r], off, 32));

#pragma unroll
        for (int r = 0; r < 8; ++r) {
            float mn    = fmaxf(m[r], rm[r]);
            float alpha = __expf(m[r] - mn);
            m[r] = mn;
            l[r] *= alpha;
#pragma unroll
            for (int dt = 0; dt < 4; ++dt) o[dt].f[r] *= alpha;
            float part = 0.f;
#pragma unroll
            for (int n = 0; n < 4; ++n) {
                float p = __expf(s[n].f[r] - mn);
                s[n].f[r] = p;
                part += p;
            }
            rs[r] = part;
        }
#pragma unroll
        for (int off = 8; off >= 1; off >>= 1)
#pragma unroll
            for (int r = 0; r < 8; ++r)
                rs[r] += __shfl_xor(rs[r], off, 32);
#pragma unroll
        for (int r = 0; r < 8; ++r) l[r] += rs[r];

        // P: C-layout regs -> LDS -> A-layout fragments
#pragma unroll
        for (int n = 0; n < 4; ++n)
#pragma unroll
            for (int r = 0; r < 8; ++r)
                Ps[wid][r + 8 * g][n * 16 + ln] = f2bf(s[n].f[r]);
        __syncthreads();

        // O += P @ V
#pragma unroll
        for (int dt = 0; dt < 4; ++dt)
#pragma unroll
            for (int ks = 0; ks < 2; ++ks) {
                FragA ap, bv;
#pragma unroll
                for (int j = 0; j < 8; ++j) {
                    int kp = ks * 32 + kpair_of(j, g);
                    ap.u[j] = *(const unsigned int*)&Ps[wid][ln][kp];
                    bv.u[j] = *(const unsigned int*)&Vt[dt * 16 + ln][kp];
                }
                o[dt].v = __builtin_amdgcn_wmma_f32_16x16x32_bf16(
                    false, ap.v, false, bv.v, (short)0, o[dt].v, false, false);
            }
    }

    // normalize and emit ctx (bf16, [B,T,E] row-major with head offset)
#pragma unroll
    for (int r = 0; r < 8; ++r) {
        float inv = 1.f / l[r];
        int row = b * SEQ + q0 + r + 8 * g;
#pragma unroll
        for (int dt = 0; dt < 4; ++dt) {
            int col = h * HDIM + dt * 16 + ln;
            ctxb[(size_t)row * EMBED + col] = f2bf(o[dt].f[r] * inv);
        }
    }
}

// ---------------------------------------------------------------------- launch
extern "C" void kernel_launch(void* const* d_in, const int* in_sizes, int n_in,
                              void* d_out, int out_size, void* d_ws, size_t ws_size,
                              hipStream_t stream) {
    (void)in_sizes; (void)n_in; (void)out_size; (void)ws_size;
    const float* x    = (const float*)d_in[0];
    const int*   mask = (const int*)  d_in[1];
    const float* Wqkv = (const float*)d_in[2];
    const float* bqkv = (const float*)d_in[3];
    const float* Wout = (const float*)d_in[4];
    const float* bout = (const float*)d_in[5];

    char* ws = (char*)d_ws;
    size_t off = 0;
    unsigned short* xb    = (unsigned short*)(ws + off); off += (size_t)ROWS  * EMBED * 2;
    unsigned short* wqkvb = (unsigned short*)(ws + off); off += (size_t)EMBED * QKV_N * 2;
    unsigned short* woutb = (unsigned short*)(ws + off); off += (size_t)EMBED * EMBED * 2;
    unsigned short* qkvb  = (unsigned short*)(ws + off); off += (size_t)ROWS  * QKV_N * 2;
    unsigned short* ctxb  = (unsigned short*)(ws + off); off += (size_t)ROWS  * EMBED * 2;

    int n1 = ROWS * EMBED;
    cvt_f32_bf16<<<(n1 + 255) / 256, 256, 0, stream>>>(x, xb, n1);
    int n2 = EMBED * QKV_N;
    cvt_f32_bf16<<<(n2 + 255) / 256, 256, 0, stream>>>(Wqkv, wqkvb, n2);
    int n3 = EMBED * EMBED;
    cvt_f32_bf16<<<(n3 + 255) / 256, 256, 0, stream>>>(Wout, woutb, n3);

    // qkv = x @ Wqkv + bqkv   (bf16 out)
    gemm_bf16_wmma<true><<<dim3(QKV_N / 128, ROWS / 128), 256, 0, stream>>>(
        xb, wqkvb, bqkv, qkvb, ROWS, QKV_N, EMBED);

    // streaming attention per (b, h, 64-query block)
    attention_wmma<<<BATCH * HEADS * (SEQ / 64), 128, 0, stream>>>(qkvb, mask, ctxb);

    // out = ctx @ Wout + bout  (fp32 out)
    gemm_bf16_wmma<false><<<dim3(EMBED / 128, ROWS / 128), 256, 0, stream>>>(
        ctxb, woutb, bout, d_out, ROWS, EMBED, EMBED);
}
